// ModelClass_61967788147286
// MI455X (gfx1250) — matrix operand order repaired and checked by reading, not compiled, not run on hardware
//
#include <hip/hip_runtime.h>

typedef __attribute__((ext_vector_type(16))) _Float16 v16h;
typedef __attribute__((ext_vector_type(8)))  float    v8f;

#define N_NODES 300000
#define P_LAYER 100000
#define L_LAYERS 3
#define E_SET   800000
#define E0_EDGES 1200000
#define G_GRAPHS 64
#define DYN  64
#define STAT 4
#define NF   68
#define KPAD 96
#define W1_ROWS 80
#define W1T_SZ (80*96)
#define W2T_SZ (64*96)
#define W3T_SZ (64*64)
#define BANK_SZ (W1T_SZ + W2T_SZ + W3T_SZ)
#define WAVES 4

// ---------------- elementwise / scatter kernels ----------------

__global__ void k_zero(float* __restrict__ p, long n) {
  long i = (long)blockIdx.x * blockDim.x + threadIdx.x;
  if (i < n) p[i] = 0.f;
}

__global__ void k_relu(float* __restrict__ p, long n) {
  long i = (long)blockIdx.x * blockDim.x + threadIdx.x;
  if (i < n) { float v = p[i]; p[i] = v > 0.f ? v : 0.f; }
}

__global__ void k_deg(const int* __restrict__ dst, int ne, float* __restrict__ deg) {
  int i = blockIdx.x * blockDim.x + threadIdx.x;
  if (i < ne) atomicAdd(&deg[dst[i]], 1.f);
}

__global__ void k_dinv(float* __restrict__ deg, int n) {
  int i = blockIdx.x * blockDim.x + threadIdx.x;
  if (i < n) deg[i] = rsqrtf(deg[i] + 1.f);   // +1 self loop; always > 0
}

__global__ void k_gcn_scatter(const int* __restrict__ src, const int* __restrict__ dst, int ne,
                              const float* __restrict__ x, const float* __restrict__ dinv,
                              float* __restrict__ t) {
  int i = blockIdx.x * blockDim.x + threadIdx.x;
  if (i < ne) {
    int s = src[i], d = dst[i];
    atomicAdd(&t[d], x[s] * dinv[s]);
  }
}

__global__ void k_gcn_expand(const float* __restrict__ x, const float* __restrict__ dinv,
                             const float* __restrict__ t, const float* __restrict__ w,
                             const float* __restrict__ b, float* __restrict__ xx) {
  long idx = (long)blockIdx.x * blockDim.x + threadIdx.x;
  if (idx >= (long)N_NODES * DYN) return;
  int i = (int)(idx >> 6), f = (int)(idx & 63);
  float s = dinv[i] * (t[i] + x[i] * dinv[i]);
  xx[idx] = s * w[f] + b[f];
}

// Pack W [K x Nout] f32 row-major -> W^T [NoutPad x Kpad] f16 zero-padded
__global__ void k_pack_wT(const float* __restrict__ w, _Float16* __restrict__ dst,
                          int K, int Nout, int Kpad, int NoutPad) {
  int idx = blockIdx.x * blockDim.x + threadIdx.x;
  if (idx >= NoutPad * Kpad) return;
  int n = idx / Kpad, k = idx - n * Kpad;
  float v = (n < Nout && k < K) ? w[k * Nout + n] : 0.f;
  dst[idx] = (_Float16)v;
}

// segment-sum of h=[xx|static][src] into agg[dst-lo]; dyn part then stat part
__global__ void k_agg(const int* __restrict__ src, const int* __restrict__ dst, int ne,
                      const float* __restrict__ xx, const float* __restrict__ st,
                      float* __restrict__ aggd, float* __restrict__ aggs, int lo) {
  long i = (long)blockIdx.x * blockDim.x + threadIdx.x;
  long dynTot = (long)ne * DYN;
  if (i < dynTot) {
    int e = (int)(i >> 6), f = (int)(i & 63);
    int s = src[e], d = dst[e];
    atomicAdd(&aggd[(long)(d - lo) * DYN + f], xx[(long)s * DYN + f]);
  } else {
    long j = i - dynTot;
    if (j < (long)ne * STAT) {
      int e = (int)(j >> 2), f = (int)(j & 3);
      int s = src[e], d = dst[e];
      atomicAdd(&aggs[(long)(d - lo) * STAT + f], st[(long)s * STAT + f]);
    }
  }
}

__global__ void k_pool(const float* __restrict__ xx, const int* __restrict__ batch,
                       float* __restrict__ pooled) {
  long idx = (long)blockIdx.x * blockDim.x + threadIdx.x;
  if (idx >= (long)N_NODES * DYN) return;
  int i = (int)(idx >> 6), f = (int)(idx & 63);
  atomicAdd(&pooled[(long)batch[i] * DYN + f], xx[idx]);
}

__global__ void k_final(const float* __restrict__ pooled, const float* __restrict__ lw,
                        const float* __restrict__ lb, float* __restrict__ out) {
  int g = threadIdx.x;
  if (g >= G_GRAPHS) return;
  float acc = lb[0];
  for (int f = 0; f < DYN; ++f) acc += pooled[g * DYN + f] * lw[f];
  out[g] = acc > 0.f ? acc : 0.f;
}

// ---------------- WMMA MLP ----------------

__device__ __forceinline__ v8f wmma_step(v16h a, v16h b, v8f c) {
  return __builtin_amdgcn_wmma_f32_16x16x32_f16(false, a, false, b, (short)0, c, false, false);
}

// A fragment 16x32 f16: lane m=lane&15; K = klo + 16*(lane>>4) + e
__device__ __forceinline__ v16h load_afrag(const _Float16* zt, int lane, int klo) {
  const _Float16* p = zt + (lane & 15) * KPAD + klo + ((lane >> 4) << 4);
  v16h a;
#pragma unroll
  for (int e = 0; e < 16; ++e) a[e] = p[e];
  return a;
}

// B fragment 32x16 f16 from transposed weights: n=lane&15; K = klo + 16*(lane>>4) + e
__device__ __forceinline__ v16h load_bfrag(const _Float16* wT, int lane, int n0, int klo, int kstride) {
  const _Float16* p = wT + (n0 + (lane & 15)) * kstride + klo + ((lane >> 4) << 4);
  v16h b;
#pragma unroll
  for (int e = 0; e < 16; ++e) b[e] = p[e];
  return b;
}

template <bool GIN>
__global__ __launch_bounds__(128) void k_mlp(
    float* __restrict__ xx, const float* __restrict__ st,
    const float* __restrict__ aggd, const float* __restrict__ aggs,
    const float* __restrict__ epsp, const _Float16* __restrict__ gW,
    const float* __restrict__ b1, const float* __restrict__ b2, const float* __restrict__ b3,
    int lo, int count) {
  __shared__ _Float16 sW[BANK_SZ];
  __shared__ _Float16 sZ[WAVES][2][16 * KPAD];

  int tid = threadIdx.x;
  for (int i = tid; i < BANK_SZ; i += 128) sW[i] = gW[i];
  __syncthreads();

  const _Float16* W1T = sW;
  const _Float16* W2T = sW + W1T_SZ;
  const _Float16* W3T = sW + W1T_SZ + W2T_SZ;

  int wave = tid >> 5, lane = tid & 31;
  int t = blockIdx.x * WAVES + wave;
  if (t * 16 >= count) return;
  long base = (long)lo + (long)t * 16;

  float mult = 1.0f + (epsp ? epsp[0] : 0.0f);
  _Float16* zA = &sZ[wave][0][0];
  _Float16* zB = &sZ[wave][1][0];

  // stage z0 = (1+eps)*[xx|static] + agg  (padded K=96); zero zB
  for (int i = lane; i < 16 * KPAD; i += 32) {
    int m = i / KPAD, k = i - m * KPAD;
    long node = base + m;
    float v = 0.f;
    if (k < DYN) {
      v = mult * xx[node * DYN + k];
      if (GIN) v += aggd[(node - lo) * DYN + k];
    } else if (k < NF) {
      v = mult * st[node * STAT + (k - DYN)];
      if (GIN) v += aggs[(node - lo) * STAT + (k - DYN)];
    }
    zA[i] = (_Float16)v;
    zB[i] = (_Float16)0.f;
  }
  asm volatile("s_wait_dscnt 0" ::: "memory");

  // ---- layer 1: [16 x 96] @ [96 x N1] -> zB ----
  {
    v16h a0 = load_afrag(zA, lane, 0);
    v16h a1 = load_afrag(zA, lane, 32);
    v16h a2 = load_afrag(zA, lane, 64);
    const int N1 = GIN ? NF : DYN;
    const int NT1 = GIN ? 5 : 4;
#pragma unroll
    for (int nt = 0; nt < NT1; ++nt) {
      int n0 = nt * 16;
      v8f acc = {};
      acc = wmma_step(a0, load_bfrag(W1T, lane, n0, 0, KPAD), acc);
      acc = wmma_step(a1, load_bfrag(W1T, lane, n0, 32, KPAD), acc);
      acc = wmma_step(a2, load_bfrag(W1T, lane, n0, 64, KPAD), acc);
      int n = n0 + (lane & 15);
      int mb = (lane >> 4) << 3;
      if (n < N1) {
        float bias = b1[n];
#pragma unroll
        for (int r = 0; r < 8; ++r) {
          float v = acc[r] + bias; v = v > 0.f ? v : 0.f;
          zB[(mb + r) * KPAD + n] = (_Float16)v;
        }
      }
    }
  }
  asm volatile("s_wait_dscnt 0" ::: "memory");

  // ---- layer 2: [16 x K2] @ [K2 x 64] -> zA (cols 0..63) ----
  {
    v16h c0 = load_afrag(zB, lane, 0);
    v16h c1 = load_afrag(zB, lane, 32);
    v16h c2 = GIN ? load_afrag(zB, lane, 64) : v16h{};
#pragma unroll
    for (int nt = 0; nt < 4; ++nt) {
      int n0 = nt * 16;
      v8f acc = {};
      acc = wmma_step(c0, load_bfrag(W2T, lane, n0, 0, KPAD), acc);
      acc = wmma_step(c1, load_bfrag(W2T, lane, n0, 32, KPAD), acc);
      if (GIN) acc = wmma_step(c2, load_bfrag(W2T, lane, n0, 64, KPAD), acc);
      int n = n0 + (lane & 15);
      int mb = (lane >> 4) << 3;
      float bias = b2[n];
#pragma unroll
      for (int r = 0; r < 8; ++r) {
        float v = acc[r] + bias; v = v > 0.f ? v : 0.f;
        zA[(mb + r) * KPAD + n] = (_Float16)v;
      }
    }
  }
  asm volatile("s_wait_dscnt 0" ::: "memory");

  // ---- layer 3: [16 x 64] @ [64 x 64] -> xx (global) ----
  {
    v16h d0 = load_afrag(zA, lane, 0);
    v16h d1 = load_afrag(zA, lane, 32);
#pragma unroll
    for (int nt = 0; nt < 4; ++nt) {
      int n0 = nt * 16;
      v8f acc = {};
      acc = wmma_step(d0, load_bfrag(W3T, lane, n0, 0, 64), acc);
      acc = wmma_step(d1, load_bfrag(W3T, lane, n0, 32, 64), acc);
      int n = n0 + (lane & 15);
      int mb = (lane >> 4) << 3;
      float bias = b3[n];
#pragma unroll
      for (int r = 0; r < 8; ++r) {
        float v = acc[r] + bias; v = v > 0.f ? v : 0.f;
        xx[(base + mb + r) * DYN + n] = v;
      }
    }
  }
}

// ---------------- host orchestration ----------------

static inline int gridFor(long n, int bs) { return (int)((n + bs - 1) / bs); }

extern "C" void kernel_launch(void* const* d_in, const int* in_sizes, int n_in,
                              void* d_out, int out_size, void* d_ws, size_t ws_size,
                              hipStream_t stream) {
  (void)in_sizes; (void)n_in; (void)out_size; (void)ws_size;
  const float* x       = (const float*)d_in[0];
  const float* statics = (const float*)d_in[1];
  const int*   eidx    = (const int*)d_in[2];
  const int*   inner   = (const int*)d_in[3];
  const int*   fwd     = (const int*)d_in[4];
  const int*   bwd     = (const int*)d_in[5];
  const int*   batch   = (const int*)d_in[7];
  const float* gcn_w   = (const float*)d_in[8];
  const float* gcn_b   = (const float*)d_in[9];
  const float* nd_w1 = (const float*)d_in[10]; const float* nd_b1 = (const float*)d_in[11];
  const float* nd_w2 = (const float*)d_in[12]; const float* nd_b2 = (const float*)d_in[13];
  const float* nd_w3 = (const float*)d_in[14]; const float* nd_b3 = (const float*)d_in[15];
  const float* lin_w = (const float*)d_in[16]; const float* lin_b = (const float*)d_in[17];
  // in_: 18..24, fw_: 25..31, bw_: 32..38  (eps,w1,b1,w2,b2,w3,b3)
  const float* eps[3]; const float* w1[3]; const float* bb1[3];
  const float* w2[3]; const float* bb2[3]; const float* w3[3]; const float* bb3[3];
  for (int b = 0; b < 3; ++b) {
    int o = 18 + 7 * b;
    eps[b] = (const float*)d_in[o + 0];
    w1[b]  = (const float*)d_in[o + 1]; bb1[b] = (const float*)d_in[o + 2];
    w2[b]  = (const float*)d_in[o + 3]; bb2[b] = (const float*)d_in[o + 4];
    w3[b]  = (const float*)d_in[o + 5]; bb3[b] = (const float*)d_in[o + 6];
  }
  float* out = (float*)d_out;

  // workspace carve-up
  char* ws = (char*)d_ws;
  auto carve = [&](size_t bytes) { void* p = (void*)ws; ws += (bytes + 255) & ~(size_t)255; return p; };
  float* xx     = (float*)carve((size_t)N_NODES * DYN * 4);
  float* aggd   = (float*)carve((size_t)P_LAYER * (DYN + STAT) * 4);  // aggs contiguous after aggd
  float* aggs   = aggd + (size_t)P_LAYER * DYN;
  float* dinv   = (float*)carve((size_t)N_NODES * 4);
  float* tsum   = (float*)carve((size_t)N_NODES * 4);
  float* pooled = (float*)carve((size_t)G_GRAPHS * DYN * 4);
  _Float16* banks = (_Float16*)carve((size_t)4 * BANK_SZ * 2);
  _Float16* bank[3] = { banks, banks + BANK_SZ, banks + 2 * BANK_SZ };   // in, fw, bw
  _Float16* bankNd = banks + 3 * BANK_SZ;

  const int BS = 256;

  // ---- pack weights (f32 -> f16 transposed, padded) ----
  for (int b = 0; b < 3; ++b) {
    k_pack_wT<<<gridFor(W1T_SZ, BS), BS, 0, stream>>>(w1[b], bank[b], NF, NF, KPAD, W1_ROWS);
    k_pack_wT<<<gridFor(W2T_SZ, BS), BS, 0, stream>>>(w2[b], bank[b] + W1T_SZ, NF, DYN, KPAD, 64);
    k_pack_wT<<<gridFor(W3T_SZ, BS), BS, 0, stream>>>(w3[b], bank[b] + W1T_SZ + W2T_SZ, DYN, DYN, 64, 64);
  }
  k_pack_wT<<<gridFor(W1T_SZ, BS), BS, 0, stream>>>(nd_w1, bankNd, NF, DYN, KPAD, W1_ROWS);
  k_pack_wT<<<gridFor(W2T_SZ, BS), BS, 0, stream>>>(nd_w2, bankNd + W1T_SZ, DYN, DYN, KPAD, 64);
  k_pack_wT<<<gridFor(W3T_SZ, BS), BS, 0, stream>>>(nd_w3, bankNd + W1T_SZ + W2T_SZ, DYN, DYN, 64, 64);

  // ---- GCN upscale 1 -> 64 ----
  k_zero<<<gridFor(N_NODES, BS), BS, 0, stream>>>(dinv, N_NODES);
  k_zero<<<gridFor(N_NODES, BS), BS, 0, stream>>>(tsum, N_NODES);
  k_deg<<<gridFor(E0_EDGES, BS), BS, 0, stream>>>(eidx + E0_EDGES, E0_EDGES, dinv);
  k_dinv<<<gridFor(N_NODES, BS), BS, 0, stream>>>(dinv, N_NODES);
  k_gcn_scatter<<<gridFor(E0_EDGES, BS), BS, 0, stream>>>(eidx, eidx + E0_EDGES, E0_EDGES, x, dinv, tsum);
  k_gcn_expand<<<gridFor((long)N_NODES * DYN, BS), BS, 0, stream>>>(x, dinv, tsum, gcn_w, gcn_b, xx);

  const long aggN = (long)P_LAYER * (DYN + STAT);
  const long aggWork = (long)E_SET * (DYN + STAT);
  const int mlpGridP = gridFor(P_LAYER / 16, WAVES);
  const int mlpGridN = gridFor(N_NODES / 16, WAVES);

  auto gin = [&](const int* srcp, const int* dstp, int b, int lo) {
    k_zero<<<gridFor(aggN, BS), BS, 0, stream>>>(aggd, aggN);
    k_agg<<<gridFor(aggWork, BS), BS, 0, stream>>>(srcp, dstp, E_SET, xx, statics, aggd, aggs, lo);
    k_mlp<true><<<mlpGridP, 128, 0, stream>>>(xx, statics, aggd, aggs, eps[b], bank[b],
                                              bb1[b], bb2[b], bb3[b], lo, P_LAYER);
  };

  const size_t ES2 = (size_t)2 * E_SET;
  for (int it = 0; it < 2; ++it) {
    // forward sweep
    for (int l = 0; l < L_LAYERS; ++l) {
      gin(inner + l * ES2, inner + l * ES2 + E_SET, 0, l * P_LAYER);
      if (l < L_LAYERS - 1)
        gin(fwd + l * ES2, fwd + l * ES2 + E_SET, 1, (l + 1) * P_LAYER);
    }
    k_relu<<<gridFor((long)N_NODES * DYN, BS), BS, 0, stream>>>(xx, (long)N_NODES * DYN);
    // backward sweep
    for (int l = L_LAYERS - 1; l >= 1; --l) {
      gin(bwd + (l - 1) * ES2, bwd + (l - 1) * ES2 + E_SET, 2, (l - 1) * P_LAYER);
      gin(inner + (l - 1) * ES2, inner + (l - 1) * ES2 + E_SET, 0, (l - 1) * P_LAYER);
    }
    k_relu<<<gridFor((long)N_NODES * DYN, BS), BS, 0, stream>>>(xx, (long)N_NODES * DYN);
    // node_dnn on all nodes
    k_mlp<false><<<mlpGridN, 128, 0, stream>>>(xx, statics, nullptr, nullptr, nullptr, bankNd,
                                               nd_b1, nd_b2, nd_b3, 0, N_NODES);
  }

  // ---- pool + final linear ----
  k_zero<<<gridFor((long)G_GRAPHS * DYN, BS), BS, 0, stream>>>(pooled, (long)G_GRAPHS * DYN);
  k_pool<<<gridFor((long)N_NODES * DYN, BS), BS, 0, stream>>>(xx, batch, pooled);
  k_final<<<1, 64, 0, stream>>>(pooled, lin_w, lin_b, out);
}
